// GATEncoder_21844203667981
// MI455X (gfx1250) — matrix operand order repaired and checked by reading, not compiled, not run on hardware
//
#include <hip/hip_runtime.h>
#include <cstdint>

#define NEG_SLOPE 0.2f
#define BN_EPS 1e-5f

typedef __attribute__((ext_vector_type(16))) __bf16 v16bf;
typedef __attribute__((ext_vector_type(8)))  float  v8f;

// ----------------------------- utility kernels ------------------------------

__global__ void k_fill_f32(float* p, float v, long n) {
  long i = (long)blockIdx.x * blockDim.x + threadIdx.x;
  if (i < n) p[i] = v;
}

__global__ void k_f32_to_bf16(const float* __restrict__ in,
                              unsigned short* __restrict__ out, long n) {
  long i = (long)blockIdx.x * blockDim.x + threadIdx.x;
  if (i < n) {
    unsigned int u = __float_as_uint(in[i]);
    out[i] = (unsigned short)((u + 0x7FFFu + ((u >> 16) & 1u)) >> 16);  // RNE
  }
}

__global__ void k_reduce_sum(const float* __restrict__ a, long n, float* out) {
  __shared__ float sm[256];
  float s = 0.f;
  for (long i = (long)blockIdx.x * blockDim.x + threadIdx.x; i < n;
       i += (long)gridDim.x * blockDim.x)
    s += a[i];
  sm[threadIdx.x] = s;
  __syncthreads();
  for (int o = 128; o > 0; o >>= 1) {
    if ((int)threadIdx.x < o) sm[threadIdx.x] += sm[threadIdx.x + o];
    __syncthreads();
  }
  if (threadIdx.x == 0) atomicAdd(out, sm[0]);
}

// --------------------------- WMMA bf16 GEMM ---------------------------------
// C[M,Nc] = A[M,K] (bf16, row major) * B[K,Nc] (bf16, row major), fp32 accum.
// One wave32 computes a 16x64 output strip. Per 32-deep k-step: the A fragment
// and ALL FOUR B fragments are loaded first (8 dword A loads, 64 u16 B loads in flight),
// then 4 independent WMMAs consume them — lets the scheduler stagger
// s_wait_loadcnt and overlap WMMA issue with outstanding loads.
// VGPR layouts per CDNA5 ISA 7.12.2 (16-bit A 16x32; B col=lane%16; C/D f32).
__global__ void k_gemm_bf16_wmma(const unsigned short* __restrict__ A,
                                 const unsigned short* __restrict__ B,
                                 float* __restrict__ C, int K, int Nc) {
  const int lane   = threadIdx.x;          // blockDim.x == 32
  const int tileM  = blockIdx.x << 4;
  const int tileN0 = blockIdx.y << 6;      // 64-wide strip
  const int m      = lane & 15;
  const int koff   = (lane >> 4) << 3;                 // 0 or 8 for A halves
  const int kb     = (lane & 15) | ((lane >> 4) << 4);

  const unsigned short* Arow = A + (size_t)(tileM + m) * K;
  v8f acc[4] = {};
  for (int k0 = 0; k0 < K; k0 += 32) {
    union { unsigned int u[8]; v16bf v; } a;
#pragma unroll
    for (int r = 0; r < 8; ++r) {
      int k = k0 + ((r & 4) ? 16 : 0) + koff + ((r & 3) << 1);
      a.u[r] = *(const unsigned int*)(Arow + k);       // two bf16 (k, k+1)
    }
    const unsigned short* Brow = B + (size_t)k0 * Nc + tileN0 + (lane & 15);
    union { uint4 q[2]; unsigned short e[16]; v16bf v; } b[4];
#pragma unroll
    for (int t = 0; t < 4; ++t) {                      // issue all B loads first
#pragma unroll
      for (int s = 0; s < 16; ++s)
        b[t].e[s] = Brow[(size_t)(koff + (s & 7) + ((s & 8) << 1)) * Nc + t * 16];
    }
#pragma unroll
    for (int t = 0; t < 4; ++t)                        // then back-to-back WMMA
      acc[t] = __builtin_amdgcn_wmma_f32_16x16x32_bf16(
          false, a.v, false, b[t].v, (short)0, acc[t], false, false);
  }
  const int n  = lane & 15;
  const int mb = tileM + ((lane >> 4) << 3);
#pragma unroll
  for (int t = 0; t < 4; ++t)
#pragma unroll
    for (int r = 0; r < 8; ++r)
      C[(size_t)(mb + r) * Nc + tileN0 + t * 16 + n] = acc[t][r];
}

// ------------------------------ edge kernels --------------------------------

__device__ inline void atomicMaxF(float* addr, float val) {
  if (val >= 0.f) atomicMax((int*)addr, __float_as_int(val));
  else            atomicMin((unsigned int*)addr, __float_as_uint(val));
}

// Wave per edge; 8 lanes per head (H==4). Computes attention logit per head,
// stores it and atomically folds the per-destination max. CC is out_channels.
template <int CC>
__global__ void k_edge_score(const float* __restrict__ xl, const float* __restrict__ xr,
                             const int* __restrict__ ei, const float* __restrict__ eattr,
                             const float* __restrict__ easum,
                             const float* __restrict__ We, const float* __restrict__ att,
                             float* __restrict__ score, float* __restrict__ emax,
                             int E, int Etot) {
  const int wave = (int)(((long)blockIdx.x * blockDim.x + threadIdx.x) >> 5);
  const int lane = threadIdx.x & 31;
  if (wave >= Etot) return;
  int s, d; float ea;
  if (wave < E) { s = ei[wave]; d = ei[(long)E + wave]; ea = eattr[wave]; }
  else          { s = d = wave - E; ea = easum[0] * (1.0f / E); }
  constexpr int PER = CC >> 3;          // channels per lane (8 lanes per head)
  constexpr int HC  = CC << 2;          // H * CC
  const int h  = lane >> 3;
  const int c0 = (lane & 7) * PER;
  const float* pl = xl + (size_t)s * HC + h * CC + c0;
  const float* pr = xr + (size_t)d * HC + h * CC + c0;
  const float* pw = We  + h * CC + c0;
  const float* pa = att + h * CC + c0;
  float p = 0.f;
#pragma unroll
  for (int j = 0; j < PER; j += 4) {
    float4 a = *(const float4*)(pl + j);
    float4 b = *(const float4*)(pr + j);
    float4 w = *(const float4*)(pw + j);
    float4 t = *(const float4*)(pa + j);
    float m0 = a.x + b.x + ea * w.x;
    float m1 = a.y + b.y + ea * w.y;
    float m2 = a.z + b.z + ea * w.z;
    float m3 = a.w + b.w + ea * w.w;
    p += (m0 > 0.f ? m0 : NEG_SLOPE * m0) * t.x;
    p += (m1 > 0.f ? m1 : NEG_SLOPE * m1) * t.y;
    p += (m2 > 0.f ? m2 : NEG_SLOPE * m2) * t.z;
    p += (m3 > 0.f ? m3 : NEG_SLOPE * m3) * t.w;
  }
  p += __shfl_xor(p, 1, 32);
  p += __shfl_xor(p, 2, 32);
  p += __shfl_xor(p, 4, 32);
  if ((lane & 7) == 0) {
    score[(size_t)wave * 4 + h] = p;
    atomicMaxF(&emax[(size_t)d * 4 + h], p);
  }
}

// One thread per (edge, head): exponentiate and fold softmax denominator.
__global__ void k_edge_exp(const int* __restrict__ ei,
                           float* __restrict__ score, const float* __restrict__ emax,
                           float* __restrict__ denom, int E, int Etot) {
  long i = (long)blockIdx.x * blockDim.x + threadIdx.x;
  if (i >= (long)Etot * 4) return;
  int e = (int)(i >> 2);
  int h = (int)(i & 3);
  int d = (e < E) ? ei[(long)E + e] : (e - E);
  float ex = expf(score[i] - emax[(size_t)d * 4 + h]);
  score[i] = ex;
  atomicAdd(&denom[(size_t)d * 4 + h], ex);
}

// Wave per edge: alpha-weighted scatter of xl[src] rows into destination.
template <int CC, int CONCAT>
__global__ void k_edge_aggr(const float* __restrict__ xl, const int* __restrict__ ei,
                            const float* __restrict__ score, const float* __restrict__ denom,
                            float* __restrict__ out, int E, int Etot) {
  const int wave = (int)(((long)blockIdx.x * blockDim.x + threadIdx.x) >> 5);
  const int lane = threadIdx.x & 31;
  if (wave >= Etot) return;
  int s, d;
  if (wave < E) { s = ei[wave]; d = ei[(long)E + wave]; }
  else          { s = d = wave - E; }
  constexpr int PER = CC >> 3;
  constexpr int HC  = CC << 2;
  const int h  = lane >> 3;
  const int c0 = (lane & 7) * PER;
  float alpha = score[(size_t)wave * 4 + h] / (denom[(size_t)d * 4 + h] + 1e-16f);
  if (!CONCAT) alpha *= 0.25f;          // mean over heads folded into scale
  const float* pl = xl + (size_t)s * HC + h * CC + c0;
  float* po = CONCAT ? (out + (size_t)d * HC + h * CC + c0)
                     : (out + (size_t)d * CC + c0);
#pragma unroll
  for (int j = 0; j < PER; j += 4) {
    float4 a = *(const float4*)(pl + j);
    atomicAdd(&po[j + 0], alpha * a.x);
    atomicAdd(&po[j + 1], alpha * a.y);
    atomicAdd(&po[j + 2], alpha * a.z);
    atomicAdd(&po[j + 3], alpha * a.w);
  }
}

// ------------------------------- epilogues ----------------------------------

__global__ void k_post_bn_elu(const float* __restrict__ agg, const float* __restrict__ bias,
                              const float* __restrict__ gamma, const float* __restrict__ beta,
                              const float* __restrict__ mean, const float* __restrict__ var,
                              float* __restrict__ out, long n, int hc) {
  long i = (long)blockIdx.x * blockDim.x + threadIdx.x;
  if (i >= n) return;
  int j = (int)(i % hc);
  float v = agg[i] + bias[j];
  v = (v - mean[j]) * rsqrtf(var[j] + BN_EPS) * gamma[j] + beta[j];
  out[i] = v > 0.f ? v : (expf(v) - 1.f);   // ELU
}

__global__ void k_add_bias(const float* __restrict__ agg, const float* __restrict__ bias,
                           float* __restrict__ out, long n, int dim) {
  long i = (long)blockIdx.x * blockDim.x + threadIdx.x;
  if (i < n) out[i] = agg[i] + bias[(int)(i % dim)];
}

// ------------------------------ host orchestration --------------------------

template <int CC, int CONCAT>
static void run_layer(const float* in, int inDim, const float* Wl, const float* Wr,
                      const float* We, const float* att,
                      const int* ei, const float* eattr, const float* easum,
                      float* xl, float* xr, float* agg, float* score,
                      float* emax, float* denom,
                      unsigned short* xbf, unsigned short* wlbf, unsigned short* wrbf,
                      hipStream_t stream) {
  const int N = 50000, E = 800000, Etot = E + N;
  const int hc = CC * 4;

  long nIn = (long)N * inDim;
  k_f32_to_bf16<<<(nIn + 255) / 256, 256, 0, stream>>>(in, xbf, nIn);
  long nW = (long)inDim * hc;
  k_f32_to_bf16<<<(nW + 255) / 256, 256, 0, stream>>>(Wl, wlbf, nW);
  k_f32_to_bf16<<<(nW + 255) / 256, 256, 0, stream>>>(Wr, wrbf, nW);

  dim3 g(N / 16, hc / 64);                 // 50000 = 3125 * 16, hc % 64 == 0
  k_gemm_bf16_wmma<<<g, 32, 0, stream>>>(xbf, wlbf, xl, inDim, hc);
  k_gemm_bf16_wmma<<<g, 32, 0, stream>>>(xbf, wrbf, xr, inDim, hc);

  long nNH = (long)N * 4;
  k_fill_f32<<<(nNH + 255) / 256, 256, 0, stream>>>(emax, -3.0e38f, nNH);
  k_fill_f32<<<(nNH + 255) / 256, 256, 0, stream>>>(denom, 0.f, nNH);
  long nOut = (long)N * (CONCAT ? hc : CC);
  k_fill_f32<<<(nOut + 255) / 256, 256, 0, stream>>>(agg, 0.f, nOut);

  int gridE = (Etot + 7) / 8;              // 8 waves per 256-thread block
  k_edge_score<CC><<<gridE, 256, 0, stream>>>(xl, xr, ei, eattr, easum, We, att,
                                              score, emax, E, Etot);
  long nSc = (long)Etot * 4;
  k_edge_exp<<<(nSc + 255) / 256, 256, 0, stream>>>(ei, score, emax, denom, E, Etot);
  k_edge_aggr<CC, CONCAT><<<gridE, 256, 0, stream>>>(xl, ei, score, denom, agg, E, Etot);
}

extern "C" void kernel_launch(void* const* d_in, const int* in_sizes, int n_in,
                              void* d_out, int out_size, void* d_ws, size_t ws_size,
                              hipStream_t stream) {
  const int N = 50000, E = 800000;
  (void)in_sizes; (void)n_in; (void)out_size; (void)ws_size;

  const float* x     = (const float*)d_in[0];
  const int*   ei    = (const int*)d_in[1];     // edge_index [2, E]
  const float* eattr = (const float*)d_in[2];
  const float* Wl0 = (const float*)d_in[3],  *Wr0 = (const float*)d_in[4];
  const float* We0 = (const float*)d_in[5],  *att0 = (const float*)d_in[6];
  const float* b0  = (const float*)d_in[7];
  const float* Wl1 = (const float*)d_in[8],  *Wr1 = (const float*)d_in[9];
  const float* We1 = (const float*)d_in[10], *att1 = (const float*)d_in[11];
  const float* b1  = (const float*)d_in[12];
  const float* Wl2 = (const float*)d_in[13], *Wr2 = (const float*)d_in[14];
  const float* We2 = (const float*)d_in[15], *att2 = (const float*)d_in[16];
  const float* b2  = (const float*)d_in[17];
  const float* g0  = (const float*)d_in[18], *be0 = (const float*)d_in[19];
  const float* m0  = (const float*)d_in[20], *v0  = (const float*)d_in[21];
  const float* g1  = (const float*)d_in[22], *be1 = (const float*)d_in[23];
  const float* m1  = (const float*)d_in[24], *v1  = (const float*)d_in[25];

  // ---- workspace carve-out (bump allocator, 256B aligned) ----
  char* ws = (char*)d_ws; size_t off = 0;
  auto take = [&](size_t bytes) -> void* {
    void* p = ws + off; off += (bytes + 255) & ~(size_t)255; return p;
  };
  float* hbuf0 = (float*)take((size_t)N * 256 * 4);
  float* hbuf1 = (float*)take((size_t)N * 256 * 4);
  float* xl    = (float*)take((size_t)N * 512 * 4);
  float* xr    = (float*)take((size_t)N * 512 * 4);
  float* agg   = (float*)take((size_t)N * 256 * 4);
  float* score = (float*)take((size_t)(E + N) * 4 * 4);
  float* emax  = (float*)take((size_t)N * 4 * 4);
  float* denom = (float*)take((size_t)N * 4 * 4);
  float* easum = (float*)take(256);
  unsigned short* xbf  = (unsigned short*)take((size_t)N * 256 * 2);
  unsigned short* wlbf = (unsigned short*)take((size_t)256 * 512 * 2);
  unsigned short* wrbf = (unsigned short*)take((size_t)256 * 512 * 2);

  // edge_attr mean (self-loop fill value)
  k_fill_f32<<<1, 256, 0, stream>>>(easum, 0.f, 1);
  k_reduce_sum<<<1024, 256, 0, stream>>>(eattr, (long)E, easum);

  // layer 0: in 128 -> 4 heads x 64, concat
  run_layer<64, 1>(x, 128, Wl0, Wr0, We0, att0, ei, eattr, easum,
                   xl, xr, agg, score, emax, denom, xbf, wlbf, wrbf, stream);
  long n256 = (long)N * 256;
  k_post_bn_elu<<<(n256 + 255) / 256, 256, 0, stream>>>(agg, b0, g0, be0, m0, v0,
                                                        hbuf0, n256, 256);
  // layer 1: in 256 -> 4 x 64, concat
  run_layer<64, 1>(hbuf0, 256, Wl1, Wr1, We1, att1, ei, eattr, easum,
                   xl, xr, agg, score, emax, denom, xbf, wlbf, wrbf, stream);
  k_post_bn_elu<<<(n256 + 255) / 256, 256, 0, stream>>>(agg, b1, g1, be1, m1, v1,
                                                        hbuf1, n256, 256);
  // layer 2: in 256 -> 4 x 128, head-mean
  run_layer<128, 0>(hbuf1, 256, Wl2, Wr2, We2, att2, ei, eattr, easum,
                    xl, xr, agg, score, emax, denom, xbf, wlbf, wrbf, stream);
  long n128 = (long)N * 128;
  k_add_bias<<<(n128 + 255) / 256, 256, 0, stream>>>(agg, b2, (float*)d_out, n128, 128);
}
